// MAB_37409165148591
// MI455X (gfx1250) — matrix operand order repaired and checked by reading, not compiled, not run on hardware
//
#include <hip/hip_runtime.h>
#include <hip/hip_bf16.h>

typedef __attribute__((ext_vector_type(16))) __bf16 v16bf;
typedef __attribute__((ext_vector_type(2)))  __bf16 v2bf;
typedef __attribute__((ext_vector_type(8)))  float  v8f;

constexpr int Bc = 4, LQc = 2048, LKc = 2048, Dc = 512, Hc = 8;

static __device__ __forceinline__ __bf16 f2bf(float f) { return (__bf16)f; }

static __device__ __forceinline__ v8f zero8() {
  v8f z;
#pragma unroll
  for (int j = 0; j < 8; ++j) z[j] = 0.0f;
  return z;
}

// Inverse of the A-fragment pattern (16x32 bf16): element (row, kk) ->
// lane = row + 16*half, slot e.
static __device__ __forceinline__ void a_pos(int kk, int& half, int& e) {
  const int kkE = kk & ~1;
  int j;
  if (kkE < 16) { half = kkE >= 8; j = (kkE - 8 * half) >> 1; }
  else { const int t = kkE - 16; half = t >= 8; j = 4 + ((t - 8 * half) >> 1); }
  e = 2 * j + (kk & 1);
}
// Inverse of the B-fragment pattern (32x16 bf16): element (kk, col) ->
// lane = col + 16*half, slot e.
static __device__ __forceinline__ void b_pos(int kk, int& half, int& e) {
  const int kkE = kk & ~1;
  int j;
  if (kkE < 16) { half = kkE >= 8; j = (kkE - 8 * half) >> 1; }
  else { const int t = kkE - 16; half = t >= 8; j = 4 + ((t - 8 * half) >> 1); }
  e = 2 * j + (kk & 1);
}

// ---------------------------------------------------------------------------
// bf16 WMMA GEMM: C[M x 512] = A[M x 512](f32) * W[512 x 512](f32) + bias
// LDS tiles held in fragment-image layout; fragment reads are 32B vector loads.
// MODE 0: Q-frag image out  [b][h][qblk][ks2][lane32][16]
// MODE 1: K-frag image out  [b][h][keyblk][ks2][lane32][16]   (B-operand of QK^T)
// MODE 2: V-frag image out  [b][key32blk][ctile32][lane32][16] (B-operand of PV)
// MODE 3: f32 row-major out with bias+ReLU
// ---------------------------------------------------------------------------
template <int MODE>
__global__ __launch_bounds__(256) void gemm_bf16_k(
    const float* __restrict__ A, const float* __restrict__ W,
    const float* __restrict__ bias, void* __restrict__ out, int M) {
  constexpr int BM = 128, BK = 32;
  __shared__ __bf16 Asf[8][32][16];  // 8 row-blocks of 16, frag image
  __shared__ __bf16 Bsf[8][32][16];  // 8 col-blocks of 16, frag image

  const int N = Dc, Kd = Dc;
  const int row0 = blockIdx.x * BM, col0 = blockIdx.y * 128;
  const int tid = threadIdx.x;
  const int lane = tid & 31, wid = tid >> 5;
  const int wm = wid & 3, wn = wid >> 2;  // wave tile: rows wm*32, cols wn*64

  v8f acc[2][4];
#pragma unroll
  for (int i = 0; i < 2; ++i)
#pragma unroll
    for (int n = 0; n < 4; ++n) acc[i][n] = zero8();

  for (int k0 = 0; k0 < Kd; k0 += BK) {
    // ---- stage A tile 128x32 into frag image (packed b32 stores) ----
    {
      const int r = tid >> 3, c4 = (tid & 7) * 4;
#pragma unroll
      for (int rr = 0; rr < BM; rr += 32) {
        const int gr = r + rr, mblk = gr >> 4, row = gr & 15;
        const float4 v = *(const float4*)(&A[(size_t)(row0 + gr) * Kd + k0 + c4]);
        const float vv[4] = {v.x, v.y, v.z, v.w};
#pragma unroll
        for (int p = 0; p < 2; ++p) {
          int half, e;
          a_pos(c4 + 2 * p, half, e);  // e is even here
          const v2bf pv = {f2bf(vv[2 * p]), f2bf(vv[2 * p + 1])};
          *(v2bf*)&Asf[mblk][row + 16 * half][e] = pv;
        }
      }
    }
    // ---- stage B tile 32x128 into frag image ----
    {
      const int rb = tid >> 5, cb4 = (tid & 31) * 4;
#pragma unroll
      for (int rr = 0; rr < BK; rr += 8) {
        const int kk = rb + rr;
        const float4 v = *(const float4*)(&W[(size_t)(k0 + kk) * N + col0 + cb4]);
        int half, e;
        b_pos(kk, half, e);
        const int nblk = cb4 >> 4, colb = (cb4 & 15) + 16 * half;
        Bsf[nblk][colb + 0][e] = f2bf(v.x);
        Bsf[nblk][colb + 1][e] = f2bf(v.y);
        Bsf[nblk][colb + 2][e] = f2bf(v.z);
        Bsf[nblk][colb + 3][e] = f2bf(v.w);
      }
    }
    __syncthreads();

    v16bf af[2], bfv[4];
#pragma unroll
    for (int i = 0; i < 2; ++i) af[i] = *(const v16bf*)&Asf[wm * 2 + i][lane][0];
#pragma unroll
    for (int n = 0; n < 4; ++n) bfv[n] = *(const v16bf*)&Bsf[wn * 4 + n][lane][0];
#pragma unroll
    for (int i = 0; i < 2; ++i)
#pragma unroll
      for (int n = 0; n < 4; ++n)
        acc[i][n] = __builtin_amdgcn_wmma_f32_16x16x32_bf16(
            false, af[i], false, bfv[n], (short)0, acc[i][n], false, false);
    __syncthreads();
  }

  // ---- epilogue ----
  const int rsub = (lane >> 4) * 8, csub = lane & 15;
#pragma unroll
  for (int i = 0; i < 2; ++i)
#pragma unroll
    for (int n = 0; n < 4; ++n)
#pragma unroll
      for (int j = 0; j < 8; ++j) {
        const int R = row0 + wm * 32 + i * 16 + rsub + j;
        const int C = col0 + wn * 64 + n * 16 + csub;
        const float v = acc[i][n][j] + bias[C];
        if (MODE == 0) {  // Q-frag image (A-operand of S)
          const int b = R >> 11, q = R & 2047, h = C >> 6, dk = C & 63;
          const int ks = dk >> 5, kk = dk & 31;
          int half, e; a_pos(kk, half, e);
          const size_t off =
              ((((size_t)(b * Hc + h) * (LQc / 16) + (q >> 4)) * 2 + ks) * 512) +
              ((q & 15) + 16 * half) * 16 + e;
          ((__bf16*)out)[off] = f2bf(v);
        } else if (MODE == 1) {  // K-frag image (B-operand of S)
          const int b = R >> 11, key = R & 2047, h = C >> 6, dk = C & 63;
          const int ks = dk >> 5, kk = dk & 31;
          int half, e; b_pos(kk, half, e);
          const size_t off =
              ((((size_t)(b * Hc + h) * (LKc / 16) + (key >> 4)) * 2 + ks) * 512) +
              ((key & 15) + 16 * half) * 16 + e;
          ((__bf16*)out)[off] = f2bf(v);
        } else if (MODE == 2) {  // V-frag image (B-operand of PV)
          const int b = R >> 11, key = R & 2047;
          const int kb32 = key >> 5, kk = key & 31;
          int half, e; b_pos(kk, half, e);
          const size_t off =
              (((size_t)(b * (LKc / 32) + kb32) * 32 + (C >> 4)) * 512) +
              ((C & 15) + 16 * half) * 16 + e;
          ((__bf16*)out)[off] = f2bf(v);
        } else {  // f32 + ReLU
          ((float*)out)[(size_t)R * N + C] = v > 0.0f ? v : 0.0f;
        }
      }
}

// ---------------------------------------------------------------------------
// Flash attention: grid (LQ/128, B*H), 256 threads (8 waves, 16 q-rows/wave)
// All matrix operands consumed as 32B fragment-image vector loads.
// ---------------------------------------------------------------------------
__global__ __launch_bounds__(256) void attn_k(
    const __bf16* __restrict__ Qfrag, const __bf16* __restrict__ Kfrag,
    const __bf16* __restrict__ Vfrag, const int* __restrict__ mask,
    float* __restrict__ O) {
  __shared__ __bf16 Plf[8][2][32][16];  // per-wave P tile, A-frag image

  const int tid = threadIdx.x, lane = tid & 31, wid = tid >> 5;
  const int bh = blockIdx.y, b = bh >> 3, h = bh & 7;
  const int qb = blockIdx.x * 8 + wid;  // 16-row q block index
  const float scale = 0.044194173824159216f;  // 1/sqrt(512)
  const int bcol = lane & 15;

  // Q fragments: one 32B load each, live whole loop
  v16bf qf[2];
  {
    const __bf16* qp =
        Qfrag + ((size_t)bh * (LQc / 16) + qb) * 2 * 512 + lane * 16;
    qf[0] = *(const v16bf*)(qp);
    qf[1] = *(const v16bf*)(qp + 512);
  }

  float m_i[8], l_i[8];
  v8f acc[4];
#pragma unroll
  for (int j = 0; j < 8; ++j) { m_i[j] = -1e30f; l_i[j] = 0.0f; }
#pragma unroll
  for (int n = 0; n < 4; ++n) acc[n] = zero8();

  const __bf16* Kf = Kfrag + (size_t)bh * (LKc / 16) * 2 * 512;
  const __bf16* Vf = Vfrag + (size_t)b * (LKc / 32) * 32 * 512;

  for (int kt = 0; kt < LKc; kt += 64) {
    // ---- S = scale * Q.K^T with pad mask ----
    v8f s[4];
#pragma unroll
    for (int n = 0; n < 4; ++n) {
      s[n] = zero8();
      const size_t kb = ((size_t)((kt >> 4) + n) * 2) * 512 + lane * 16;
#pragma unroll
      for (int ks = 0; ks < 2; ++ks) {
        const v16bf kv = *(const v16bf*)(Kf + kb + ks * 512);
        s[n] = __builtin_amdgcn_wmma_f32_16x16x32_bf16(
            false, qf[ks], false, kv, (short)0, s[n], false, false);
      }
      const int mk = mask[b * LKc + kt + n * 16 + bcol];
#pragma unroll
      for (int j = 0; j < 8; ++j) s[n][j] = mk ? s[n][j] * scale : -1e30f;
    }

    // ---- online softmax (rows live in 16-lane halves) ----
    float mnew[8], scj[8];
#pragma unroll
    for (int j = 0; j < 8; ++j) {
      float t = fmaxf(fmaxf(s[0][j], s[1][j]), fmaxf(s[2][j], s[3][j]));
      t = fmaxf(t, __shfl_xor(t, 1, 16));
      t = fmaxf(t, __shfl_xor(t, 2, 16));
      t = fmaxf(t, __shfl_xor(t, 4, 16));
      t = fmaxf(t, __shfl_xor(t, 8, 16));
      mnew[j] = fmaxf(m_i[j], t);
    }
#pragma unroll
    for (int n = 0; n < 4; ++n)
#pragma unroll
      for (int j = 0; j < 8; ++j)
        s[n][j] = (s[n][j] <= -1e29f) ? 0.0f : __expf(s[n][j] - mnew[j]);
#pragma unroll
    for (int j = 0; j < 8; ++j) {
      float rs = s[0][j] + s[1][j] + s[2][j] + s[3][j];
      rs += __shfl_xor(rs, 1, 16);
      rs += __shfl_xor(rs, 2, 16);
      rs += __shfl_xor(rs, 4, 16);
      rs += __shfl_xor(rs, 8, 16);
      scj[j] = __expf(m_i[j] - mnew[j]);
      l_i[j] = l_i[j] * scj[j] + rs;
      m_i[j] = mnew[j];
    }
#pragma unroll
    for (int n = 0; n < 4; ++n)
#pragma unroll
      for (int j = 0; j < 8; ++j) acc[n][j] *= scj[j];

    // ---- scatter P into A-frag image in LDS ----
    {
      const int srow0 = (lane >> 4) * 8;
#pragma unroll
      for (int n = 0; n < 4; ++n)
#pragma unroll
        for (int j = 0; j < 8; ++j) {
          const int key = n * 16 + bcol;       // 0..63
          const int ks = key >> 5, kk = key & 31;
          int half, e; a_pos(kk, half, e);
          Plf[wid][ks][(srow0 + j) + 16 * half][e] = f2bf(s[n][j]);
        }
    }
    __syncthreads();

    v16bf pf[2];
    pf[0] = *(const v16bf*)&Plf[wid][0][lane][0];
    pf[1] = *(const v16bf*)&Plf[wid][1][lane][0];

    // ---- O += P.V ----
#pragma unroll
    for (int ks = 0; ks < 2; ++ks) {
      const size_t vb =
          ((size_t)((kt >> 5) + ks) * 32 + h * 4) * 512 + lane * 16;
#pragma unroll
      for (int n = 0; n < 4; ++n) {
        const v16bf vv = *(const v16bf*)(Vf + vb + (size_t)n * 512);
        acc[n] = __builtin_amdgcn_wmma_f32_16x16x32_bf16(
            false, pf[ks], false, vv, (short)0, acc[n], false, false);
      }
    }
    __syncthreads();
  }

  // ---- normalize and store ----
  const int rsub = (lane >> 4) * 8;
#pragma unroll
  for (int j = 0; j < 8; ++j) {
    const float inv = l_i[j] > 0.0f ? 1.0f / l_i[j] : 0.0f;
    const size_t rowoff = ((size_t)b * LQc + qb * 16 + rsub + j) * Dc + h * 64;
#pragma unroll
    for (int n = 0; n < 4; ++n)
      O[rowoff + n * 16 + bcol] = acc[n][j] * inv;
  }
}

// ---------------------------------------------------------------------------
// LayerNorm over D=512 with fused residual: out = LN(A + R) * g + b
// One wave per row. Block 256 = 8 rows.
// ---------------------------------------------------------------------------
__global__ __launch_bounds__(256) void ln_k(
    const float* __restrict__ A, const float* __restrict__ R,
    const float* __restrict__ g, const float* __restrict__ bt,
    float* __restrict__ out) {
  const int tid = threadIdx.x, lane = tid & 31, wid = tid >> 5;
  const int row = blockIdx.x * 8 + wid;
  const float* a = A + (size_t)row * Dc;
  const float* r = R + (size_t)row * Dc;

  float x[16], s = 0.0f, s2 = 0.0f;
#pragma unroll
  for (int i = 0; i < 16; ++i) {
    const float v = a[lane + i * 32] + r[lane + i * 32];
    x[i] = v; s += v; s2 += v * v;
  }
#pragma unroll
  for (int off = 16; off > 0; off >>= 1) {
    s  += __shfl_xor(s,  off, 32);
    s2 += __shfl_xor(s2, off, 32);
  }
  const float mu = s * (1.0f / Dc);
  const float var = s2 * (1.0f / Dc) - mu * mu;
  const float rsq = rsqrtf(var + 1e-5f);
#pragma unroll
  for (int i = 0; i < 16; ++i) {
    const int c = lane + i * 32;
    out[(size_t)row * Dc + c] = (x[i] - mu) * rsq * g[c] + bt[c];
  }
}

// ---------------------------------------------------------------------------
extern "C" void kernel_launch(void* const* d_in, const int* in_sizes, int n_in,
                              void* d_out, int out_size, void* d_ws, size_t ws_size,
                              hipStream_t stream) {
  const float* Q    = (const float*)d_in[0];
  const float* K    = (const float*)d_in[1];
  const int*   mask = (const int*)d_in[2];
  const float* Wq = (const float*)d_in[3];
  const float* bq = (const float*)d_in[4];
  const float* Wk = (const float*)d_in[5];
  const float* bk = (const float*)d_in[6];
  const float* Wv = (const float*)d_in[7];
  const float* bv = (const float*)d_in[8];
  const float* Wo = (const float*)d_in[9];
  const float* bo = (const float*)d_in[10];
  const float* g0 = (const float*)d_in[11];
  const float* b0 = (const float*)d_in[12];
  const float* g1 = (const float*)d_in[13];
  const float* b1 = (const float*)d_in[14];

  char* ws = (char*)d_ws;
  __bf16* Qfrag = (__bf16*)(ws);                        //  8 MiB
  __bf16* Kfrag = (__bf16*)(ws + ((size_t)8  << 20));   //  8 MiB
  __bf16* Vfrag = (__bf16*)(ws + ((size_t)16 << 20));   //  8 MiB
  float*  attnO = (float*) (ws + ((size_t)24 << 20));   // 16 MiB
  float*  X     = (float*) (ws + ((size_t)40 << 20));   // 16 MiB
  float*  Hb    = (float*) (ws + ((size_t)56 << 20));   // 16 MiB

  const int M = Bc * LQc;  // 8192
  dim3 gg(M / 128, Dc / 128);

  gemm_bf16_k<0><<<gg, 256, 0, stream>>>(Q, Wq, bq, (void*)Qfrag, M);
  gemm_bf16_k<1><<<gg, 256, 0, stream>>>(K, Wk, bk, (void*)Kfrag, M);
  gemm_bf16_k<2><<<gg, 256, 0, stream>>>(K, Wv, bv, (void*)Vfrag, M);
  attn_k<<<dim3(LQc / 128, Bc * Hc), 256, 0, stream>>>(Qfrag, Kfrag, Vfrag, mask, attnO);
  ln_k<<<M / 8, 256, 0, stream>>>(attnO, Q, g0, b0, X);
  gemm_bf16_k<3><<<gg, 256, 0, stream>>>(X, Wo, bo, (void*)Hb, M);
  ln_k<<<M / 8, 256, 0, stream>>>(X, Hb, g1, b1, (float*)d_out);
}
